// DRL4TSP_41832981463215
// MI455X (gfx1250) — matrix operand order repaired and checked by reading, not compiled, not run on hardware
//
#include <hip/hip_runtime.h>
#include <cstdint>

// Problem constants (from reference)
#define BB 1024
#define SS 2
#define DD 1
#define HH 128
#define NN 128
#define H3 384
#define BT 16            // batch tile per decode workgroup
#define DEC_THREADS 1024 // 32 wave32s per block
#define NWAVE (DEC_THREADS / 32)

typedef __attribute__((ext_vector_type(16))) __bf16 v16bf;
typedef __attribute__((ext_vector_type(8)))  float v8f;

union FragU { v16bf v; uint4 q[2]; };
union BF8U { uint4 q; __bf16 b[8]; };

// ---------------------------------------------------------------------------
// Fragment loaders per CDNA5 16-bit WMMA lane layouts (CDNA5 ISA)
// A (16x32 bf16): lanes 0-15 row M=lane, K = [kb..kb+7] U [kb+16..kb+23], kb=0
//                 lanes 16-31 same rows, kb=8.
// B (32x16 bf16): lanes 0-15 col N=lane, K = [kb..kb+7] U [kb+16..kb+23], kb=0;
//                 lanes 16-31 same cols, kb=8.
// ---------------------------------------------------------------------------
__device__ __forceinline__ v16bf load_a16(const __bf16* base, int ld, int m0,
                                          int k0, int lane) {
  int r  = m0 + (lane & 15);
  int kb = k0 + ((lane >> 4) << 3);
  FragU f;
  const __bf16* p = base + r * ld + kb;
  f.q[0] = *(const uint4*)(p);
  f.q[1] = *(const uint4*)(p + 16);
  return f.v;
}

__device__ __forceinline__ v16bf load_b16(const __bf16* base, int ld, int n0,
                                          int k0, int lane) {
  int c  = n0 + (lane & 15);
  int kb = k0 + ((lane >> 4) << 3);
  FragU f;
  const __bf16* p = base + c * ld + kb;
  f.q[0] = *(const uint4*)(p);
  f.q[1] = *(const uint4*)(p + 16);
  return f.v;
}

__device__ __forceinline__ v8f wmma_bf16(v16bf a, v16bf b, v8f c) {
  return __builtin_amdgcn_wmma_f32_16x16x32_bf16(false, a, false, b,
                                                 (short)0, c, false, false);
}

// gfx1250 has a native V_TANH_F32 transcendental (co-executes with WMMA).
#if defined(__has_builtin)
#if __has_builtin(__builtin_amdgcn_tanhf)
#define HW_TANH(x) __builtin_amdgcn_tanhf(x)
#elif __has_builtin(__builtin_amdgcn_tanh_f32)
#define HW_TANH(x) __builtin_amdgcn_tanh_f32(x)
#endif
#endif

__device__ __forceinline__ float fast_tanh(float x) {
#ifdef HW_TANH
  return HW_TANH(x);
#else
  x = fminf(fmaxf(x, -15.f), 15.f);
  float e = __expf(2.f * x);
  return (e - 1.f) / (e + 1.f);
#endif
}
__device__ __forceinline__ float fast_sigmoid(float x) {
#ifdef HW_TANH
  return 0.5f * (1.f + HW_TANH(0.5f * x));
#else
  x = fminf(fmaxf(x, -30.f), 30.f);
  return 1.f / (1.f + __expf(-x));
#endif
}

// wave32 reductions
__device__ __forceinline__ float wave_max(float v) {
#pragma unroll
  for (int o = 16; o; o >>= 1) v = fmaxf(v, __shfl_xor(v, o, 32));
  return v;
}
__device__ __forceinline__ float wave_sum(float v) {
#pragma unroll
  for (int o = 16; o; o >>= 1) v += __shfl_xor(v, o, 32);
  return v;
}

// ---------------------------------------------------------------------------
// Kernel A: encoders (pointwise conv, S=2 / D=1) -> enc_cat bf16 [B][N][256]
// ---------------------------------------------------------------------------
__global__ void drl_encode(const float* __restrict__ st,
                           const float* __restrict__ dy,
                           const float* __restrict__ esw,
                           const float* __restrict__ esb,
                           const float* __restrict__ edw,
                           const float* __restrict__ edb,
                           __bf16* __restrict__ cat) {
  int t = blockIdx.x * blockDim.x + threadIdx.x;
  if (t >= BB * NN) return;
  int b = t >> 7, n = t & 127;
  float s0 = st[((size_t)b * SS + 0) * NN + n];
  float s1 = st[((size_t)b * SS + 1) * NN + n];
  float d0 = dy[(size_t)b * NN + n];
  __bf16* c = cat + ((size_t)b * NN + n) * 256;
#pragma unroll 4
  for (int h = 0; h < HH; ++h) {
    float se = esw[h * 2 + 0] * s0 + esw[h * 2 + 1] * s1 + esb[h];
    float de = edw[h] * d0 + edb[h];
    c[h]       = (__bf16)se;
    c[128 + h] = (__bf16)de;
  }
}

// ---------------------------------------------------------------------------
// Kernel B: convert / slice weights to bf16 once.
// ---------------------------------------------------------------------------
__global__ void drl_wconv(const float* __restrict__ w_ih,
                          const float* __restrict__ w_hh,
                          const float* __restrict__ attn_W,
                          const float* __restrict__ dec_W,
                          __bf16* wih, __bf16* whh, __bf16* Wh, __bf16* Vc,
                          __bf16* Wsd, __bf16* Vs) {
  int t = blockIdx.x * blockDim.x + threadIdx.x;
  const int n_wih = H3 * HH, n_whh = H3 * HH, n_wh = HH * HH, n_vc = HH * HH;
  const int n_wsd = HH * 256, n_vs = HH * HH;
  if (t < n_wih) { wih[t] = (__bf16)w_ih[t]; return; }
  t -= n_wih;
  if (t < n_whh) { whh[t] = (__bf16)w_hh[t]; return; }
  t -= n_whh;
  if (t < n_wh) { int h = t >> 7, k = t & 127;
    Wh[t] = (__bf16)attn_W[h * H3 + 256 + k]; return; }
  t -= n_wh;
  if (t < n_vc) { int h = t >> 7, k = t & 127;
    Vc[t] = (__bf16)dec_W[h * 2 * HH + 128 + k]; return; }
  t -= n_vc;
  if (t < n_wsd) { int h = t >> 8, k = t & 255;
    Wsd[t] = (__bf16)attn_W[h * H3 + k]; return; }
  t -= n_wsd;
  if (t < n_vs) { int h = t >> 7, k = t & 127;
    Vs[t] = (__bf16)dec_W[h * 2 * HH + k]; }
}

// ---------------------------------------------------------------------------
// Kernel C: step-invariant precompute (WMMA), one workgroup per batch row.
//   pattn[b][n][h] = sum_k Wsd[h][k] * cat[b][n][k]      (K = 256)
//   pdec [b][n][h] = sum_k Vs [h][k] * cat[b][n][k]      (K = 128, static)
// ---------------------------------------------------------------------------
__global__ __launch_bounds__(256)
void drl_precomp(const __bf16* __restrict__ cat,
                 const __bf16* __restrict__ Wsd,
                 const __bf16* __restrict__ Vs,
                 __bf16* __restrict__ pattn,
                 __bf16* __restrict__ pdec) {
  int b = blockIdx.x;
  int lane = threadIdx.x & 31, wv = threadIdx.x >> 5;
  const __bf16* catb = cat + (size_t)b * NN * 256;

  for (int t = wv; t < 128; t += 8) {
    int id = (t < 64) ? t : (t - 64);
    int m0 = (id >> 3) * 16;   // h tile
    int n0 = (id & 7) * 16;    // node tile
    v8f acc = {0.f, 0.f, 0.f, 0.f, 0.f, 0.f, 0.f, 0.f};
    if (t < 64) {
#pragma unroll
      for (int kk = 0; kk < 8; ++kk) {
        v16bf a = load_a16(Wsd, 256, m0, kk * 32, lane);
        v16bf bm = load_b16(catb, 256, n0, kk * 32, lane);
        acc = wmma_bf16(a, bm, acc);
      }
      int rb = (lane >> 4) * 8, nc = n0 + (lane & 15);
      __bf16* dst = pattn + ((size_t)b * NN + nc) * HH;
#pragma unroll
      for (int v = 0; v < 8; ++v) dst[m0 + rb + v] = (__bf16)acc[v];
    } else {
#pragma unroll
      for (int kk = 0; kk < 4; ++kk) {
        v16bf a = load_a16(Vs, 128, m0, kk * 32, lane);
        v16bf bm = load_b16(catb, 256, n0, kk * 32, lane);
        acc = wmma_bf16(a, bm, acc);
      }
      int rb = (lane >> 4) * 8, nc = n0 + (lane & 15);
      __bf16* dst = pdec + ((size_t)b * NN + nc) * HH;
#pragma unroll
      for (int v = 0; v < 8; ++v) dst[m0 + rb + v] = (__bf16)acc[v];
    }
  }
}

// ---------------------------------------------------------------------------
// Kernel D: persistent decoder. One block (32 waves) owns BT=16 batch rows
// for all N=128 steps. WMMA for GRU/projections; native v_tanh for the
// streaming score passes; wave-parallel softmax/argmax.
// ---------------------------------------------------------------------------
__global__ __launch_bounds__(DEC_THREADS)
void drl_decode(const float* __restrict__ g_static,
                const float* __restrict__ x0,
                const float* __restrict__ emb_w, const float* __restrict__ emb_b,
                const float* __restrict__ b_ih,  const float* __restrict__ b_hh,
                const float* __restrict__ attn_v, const float* __restrict__ dec_v,
                const __bf16* __restrict__ g_wih, const __bf16* __restrict__ g_whh,
                const __bf16* __restrict__ g_Wh,  const __bf16* __restrict__ g_Vc,
                const __bf16* __restrict__ g_cat,
                const __bf16* __restrict__ g_pattn,
                const __bf16* __restrict__ g_pdec,
                float* __restrict__ d_out) {
  __shared__ __align__(16) __bf16 sh_hbf[BT * HH];
  __shared__ __align__(16) __bf16 sh_embbf[BT * HH];
  __shared__ __align__(16) __bf16 sh_ctxbf[BT * HH];
  __shared__ float sh_h[BT * HH];
  __shared__ float sh_gi[BT * H3];
  __shared__ float sh_gh[BT * H3];
  __shared__ float sh_attnh[BT * HH];
  __shared__ float sh_ctxh[BT * HH];
  __shared__ float sh_sc[BT * NN];
  __shared__ float sh_embw[HH * 2];
  __shared__ float sh_embb[HH];
  __shared__ float sh_bih[H3];
  __shared__ float sh_bhh[H3];
  __shared__ float sh_av[HH];
  __shared__ float sh_dv[HH];
  __shared__ float sh_last[BT * 2];
  __shared__ float sh_logp[BT];
  __shared__ float sh_inv[BT];

  const int tid = threadIdx.x;
  const int lane = tid & 31, wv = tid >> 5;
  const int b0 = blockIdx.x * BT;

  for (int i = tid; i < HH * 2; i += DEC_THREADS) sh_embw[i] = emb_w[i];
  for (int i = tid; i < HH; i += DEC_THREADS) {
    sh_embb[i] = emb_b[i]; sh_av[i] = attn_v[i]; sh_dv[i] = dec_v[i];
  }
  for (int i = tid; i < H3; i += DEC_THREADS) {
    sh_bih[i] = b_ih[i]; sh_bhh[i] = b_hh[i];
  }
  for (int i = tid; i < BT * HH; i += DEC_THREADS) {
    sh_h[i] = 0.f; sh_hbf[i] = (__bf16)0.f;
  }
  for (int i = tid; i < BT * 2; i += DEC_THREADS) sh_last[i] = x0[i & 1];
  if (tid < BT) sh_logp[tid] = 0.f;
  __syncthreads();

  for (int step = 0; step < NN; ++step) {
    // (a) embedding: emb = last_out @ emb_w.T + emb_b  ([16,2]x[2,128])
    for (int i = tid; i < BT * HH; i += DEC_THREADS) {
      int r = i >> 7, h = i & 127;
      float e = sh_embw[h * 2] * sh_last[r * 2] +
                sh_embw[h * 2 + 1] * sh_last[r * 2 + 1] + sh_embb[h];
      sh_embbf[i] = (__bf16)e;
    }
    __syncthreads();

    // (b) GRU gemms: gi = emb@w_ih^T, gh = h@w_hh^T  (WMMA, 48 tiles)
    for (int t = wv; t < 48; t += NWAVE) {
      bool isg = t < 24;
      int j0 = (isg ? t : t - 24) * 16;
      const __bf16* A = isg ? sh_embbf : sh_hbf;
      const __bf16* Bw = isg ? g_wih : g_whh;
      v8f acc = {0.f, 0.f, 0.f, 0.f, 0.f, 0.f, 0.f, 0.f};
#pragma unroll
      for (int kk = 0; kk < 4; ++kk) {
        v16bf a = load_a16(A, HH, 0, kk * 32, lane);
        v16bf bm = load_b16(Bw, HH, j0, kk * 32, lane);
        acc = wmma_bf16(a, bm, acc);
      }
      int rb = (lane >> 4) * 8, nc = j0 + (lane & 15);
      float* dst = isg ? sh_gi : sh_gh;
#pragma unroll
      for (int v = 0; v < 8; ++v) dst[(rb + v) * H3 + nc] = acc[v];
    }
    __syncthreads();

    // (c) GRU gates (native tanh/sigmoid)
    for (int i = tid; i < BT * HH; i += DEC_THREADS) {
      int r = i >> 7, h = i & 127, base = r * H3;
      float ir = sh_gi[base + h] + sh_bih[h];
      float hr = sh_gh[base + h] + sh_bhh[h];
      float iz = sh_gi[base + 128 + h] + sh_bih[128 + h];
      float hz = sh_gh[base + 128 + h] + sh_bhh[128 + h];
      float in_ = sh_gi[base + 256 + h] + sh_bih[256 + h];
      float hn = sh_gh[base + 256 + h] + sh_bhh[256 + h];
      float rg = fast_sigmoid(ir + hr);
      float z  = fast_sigmoid(iz + hz);
      float nn = fast_tanh(in_ + rg * hn);
      float hv = (1.f - z) * nn + z * sh_h[i];
      sh_h[i] = hv; sh_hbf[i] = (__bf16)hv;
    }
    __syncthreads();

    // (d) attn_h = h @ Wh^T (WMMA, 8 tiles)
    for (int t = wv; t < 8; t += NWAVE) {
      int j0 = t * 16;
      v8f acc = {0.f, 0.f, 0.f, 0.f, 0.f, 0.f, 0.f, 0.f};
#pragma unroll
      for (int kk = 0; kk < 4; ++kk) {
        v16bf a = load_a16(sh_hbf, HH, 0, kk * 32, lane);
        v16bf bm = load_b16(g_Wh, HH, j0, kk * 32, lane);
        acc = wmma_bf16(a, bm, acc);
      }
      int rb = (lane >> 4) * 8, nc = j0 + (lane & 15);
#pragma unroll
      for (int v = 0; v < 8; ++v) sh_attnh[(rb + v) * HH + nc] = acc[v];
    }
    __syncthreads();

    // (e) attention scores: sc[r][n] = attn_v . tanh(pattn[b][n][:] + attn_h[r][:])
    for (int i = tid; i < BT * NN; i += DEC_THREADS) {
      int r = i >> 7, n = i & 127;
      const __bf16* pb = g_pattn + ((size_t)(b0 + r) * NN + n) * HH;
      if (i + DEC_THREADS < BT * NN) {
        int i2 = i + DEC_THREADS;
        __builtin_prefetch(
            g_pattn + ((size_t)(b0 + (i2 >> 7)) * NN + (i2 & 127)) * HH, 0, 0);
      }
      float acc = 0.f;
      for (int hc = 0; hc < HH; hc += 8) {
        BF8U p; p.q = *(const uint4*)(pb + hc);
#pragma unroll
        for (int ii = 0; ii < 8; ++ii) {
          float u = (float)p.b[ii] + sh_attnh[r * HH + hc + ii];
          acc += sh_av[hc + ii] * fast_tanh(u);
        }
      }
      sh_sc[i] = acc;
    }
    __syncthreads();

    // (f) softmax over nodes: one wave per row, shuffle reductions
    if (wv < BT) {
      int r = wv;
      float vals[4]; float m = -1e30f;
#pragma unroll
      for (int c = 0; c < 4; ++c) {
        vals[c] = sh_sc[r * NN + c * 32 + lane];
        m = fmaxf(m, vals[c]);
      }
      m = wave_max(m);
      float s = 0.f;
#pragma unroll
      for (int c = 0; c < 4; ++c) {
        float e = __expf(vals[c] - m);
        sh_sc[r * NN + c * 32 + lane] = e;
        s += e;
      }
      s = wave_sum(s);
      if (lane == 0) sh_inv[r] = 1.f / s;
    }
    __syncthreads();

    // (g) context[r][h] = sum_n attn[r][n] * static_enc[b][n][h]
    //     2 waves per row, each covers a 64-wide h half.
    {
      int r = wv >> 1, half = wv & 1;
      float a0 = 0.f, a1 = 0.f;
      const __bf16* cb = g_cat + (size_t)(b0 + r) * NN * 256 + half * 64;
      for (int n = 0; n < NN; ++n) {
        float a = sh_sc[r * NN + n];
        const __bf16* sp = cb + n * 256;
        a0 += a * (float)sp[lane];
        a1 += a * (float)sp[32 + lane];
      }
      float inv = sh_inv[r];
      sh_ctxbf[r * HH + half * 64 + lane]      = (__bf16)(a0 * inv);
      sh_ctxbf[r * HH + half * 64 + 32 + lane] = (__bf16)(a1 * inv);
    }
    __syncthreads();

    // (h) ctx_h = context @ Vc^T (WMMA, 8 tiles)
    for (int t = wv; t < 8; t += NWAVE) {
      int j0 = t * 16;
      v8f acc = {0.f, 0.f, 0.f, 0.f, 0.f, 0.f, 0.f, 0.f};
#pragma unroll
      for (int kk = 0; kk < 4; ++kk) {
        v16bf a = load_a16(sh_ctxbf, HH, 0, kk * 32, lane);
        v16bf bm = load_b16(g_Vc, HH, j0, kk * 32, lane);
        acc = wmma_bf16(a, bm, acc);
      }
      int rb = (lane >> 4) * 8, nc = j0 + (lane & 15);
#pragma unroll
      for (int v = 0; v < 8; ++v) sh_ctxh[(rb + v) * HH + nc] = acc[v];
    }
    __syncthreads();

    // (i) pointer logits: dec_v . tanh(pdec[b][n][:] + ctx_h[r][:])
    for (int i = tid; i < BT * NN; i += DEC_THREADS) {
      int r = i >> 7, n = i & 127;
      const __bf16* pb = g_pdec + ((size_t)(b0 + r) * NN + n) * HH;
      if (i + DEC_THREADS < BT * NN) {
        int i2 = i + DEC_THREADS;
        __builtin_prefetch(
            g_pdec + ((size_t)(b0 + (i2 >> 7)) * NN + (i2 & 127)) * HH, 0, 0);
      }
      float acc = 0.f;
      for (int hc = 0; hc < HH; hc += 8) {
        BF8U p; p.q = *(const uint4*)(pb + hc);
#pragma unroll
        for (int ii = 0; ii < 8; ++ii) {
          float u = (float)p.b[ii] + sh_ctxh[r * HH + hc + ii];
          acc += sh_dv[hc + ii] * fast_tanh(u);
        }
      }
      sh_sc[i] = acc;
    }
    __syncthreads();

    // (j) argmax / logp / gather next input: one wave per row
    if (wv < BT) {
      int r = wv;
      float vals[4]; float m = -1e30f; int am = 0;
#pragma unroll
      for (int c = 0; c < 4; ++c) {
        int n = c * 32 + lane;
        float v = sh_sc[r * NN + n];
        vals[c] = v;
        if (v > m) { m = v; am = n; }
      }
#pragma unroll
      for (int o = 16; o; o >>= 1) {
        float om = __shfl_xor(m, o, 32);
        int oi = __shfl_xor(am, o, 32);
        if (om > m || (om == m && oi < am)) { m = om; am = oi; }
      }
      float s = 0.f;
#pragma unroll
      for (int c = 0; c < 4; ++c) s += __expf(vals[c] - m);
      s = wave_sum(s);
      if (lane == 0) {
        sh_logp[r] -= __logf(s);  // log(max prob) = -log(sum exp(l - lmax))
        sh_last[r * 2 + 0] = g_static[((size_t)(b0 + r) * SS + 0) * NN + am];
        sh_last[r * 2 + 1] = g_static[((size_t)(b0 + r) * SS + 1) * NN + am];
        d_out[(size_t)(b0 + r) * NN + step] = (float)am;  // tour_idx
      }
    }
    __syncthreads();
  }

  if (tid < BT) d_out[(size_t)BB * NN + b0 + tid] = sh_logp[tid];  // tour_logp
}

// ---------------------------------------------------------------------------
extern "C" void kernel_launch(void* const* d_in, const int* in_sizes, int n_in,
                              void* d_out, int out_size, void* d_ws, size_t ws_size,
                              hipStream_t stream) {
  (void)in_sizes; (void)n_in; (void)out_size; (void)ws_size;
  const float* st     = (const float*)d_in[0];
  const float* dy     = (const float*)d_in[1];
  const float* esw    = (const float*)d_in[2];
  const float* esb    = (const float*)d_in[3];
  const float* edw    = (const float*)d_in[4];
  const float* edb    = (const float*)d_in[5];
  const float* x0     = (const float*)d_in[6];
  const float* emb_w  = (const float*)d_in[7];
  const float* emb_b  = (const float*)d_in[8];
  const float* w_ih   = (const float*)d_in[9];
  const float* w_hh   = (const float*)d_in[10];
  const float* b_ih   = (const float*)d_in[11];
  const float* b_hh   = (const float*)d_in[12];
  const float* attn_v = (const float*)d_in[13];
  const float* attn_W = (const float*)d_in[14];
  const float* dec_v  = (const float*)d_in[15];
  const float* dec_W  = (const float*)d_in[16];
  float* out = (float*)d_out;

  // workspace carve-up (all bf16, 256B aligned); total ~135 MB (< 192MB L2)
  uint8_t* ws = (uint8_t*)d_ws;
  size_t off = 0;
  auto carve = [&](size_t bytes) {
    uint8_t* p = ws + off;
    off = (off + bytes + 255) & ~(size_t)255;
    return p;
  };
  __bf16* cat   = (__bf16*)carve((size_t)BB * NN * 256 * 2);
  __bf16* pattn = (__bf16*)carve((size_t)BB * NN * HH * 2);
  __bf16* pdec  = (__bf16*)carve((size_t)BB * NN * HH * 2);
  __bf16* wih   = (__bf16*)carve((size_t)H3 * HH * 2);
  __bf16* whh   = (__bf16*)carve((size_t)H3 * HH * 2);
  __bf16* Wh    = (__bf16*)carve((size_t)HH * HH * 2);
  __bf16* Vc    = (__bf16*)carve((size_t)HH * HH * 2);
  __bf16* Wsd   = (__bf16*)carve((size_t)HH * 256 * 2);
  __bf16* Vs    = (__bf16*)carve((size_t)HH * HH * 2);

  // A: encoders -> enc_cat bf16
  drl_encode<<<(BB * NN + 255) / 256, 256, 0, stream>>>(st, dy, esw, esb, edw,
                                                        edb, cat);
  // B: weight conversion/slicing
  {
    int total = H3 * HH * 2 + HH * HH * 2 + HH * 256 + HH * HH;
    drl_wconv<<<(total + 255) / 256, 256, 0, stream>>>(w_ih, w_hh, attn_W,
                                                       dec_W, wih, whh, Wh, Vc,
                                                       Wsd, Vs);
  }
  // C: step-invariant WMMA precompute (one block per batch row)
  drl_precomp<<<BB, 256, 0, stream>>>(cat, Wsd, Vs, pattn, pdec);
  // D: persistent decode, 64 blocks x 32 waves
  drl_decode<<<BB / BT, DEC_THREADS, 0, stream>>>(st, x0, emb_w, emb_b, b_ih,
                                                  b_hh, attn_v, dec_v, wih,
                                                  whh, Wh, Vc, cat, pattn,
                                                  pdec, out);
}
